// LongformerSelfAttention_54958401519621
// MI455X (gfx1250) — matrix operand order repaired and checked by reading, not compiled, not run on hardware
//
#include <hip/hip_runtime.h>
#include <hip/hip_bf16.h>
#include <stdint.h>

// Problem constants (match reference)
constexpr int Bn = 2, Sn = 4096, En = 768, Hn = 12, Dn = 64, Wn = 256;
constexpr int Cn = Sn / Wn;            // 16 query chunks of 256
constexpr int KT = (3 * Wn) / 32;      // 24 key tiles of 32 within the 3W window

typedef __attribute__((ext_vector_type(16))) __bf16 v16bf;
typedef __attribute__((ext_vector_type(8)))  float  v8f;

union FragAB { uint32_t u[8]; uint4 q[2]; v16bf v; };
union FragC  { float f[8]; v8f v; };

__device__ __forceinline__ uint16_t f2bf(float x) {
  uint32_t u = __float_as_uint(x);
  u += 0x7FFFu + ((u >> 16) & 1u);       // round-to-nearest-even
  return (uint16_t)(u >> 16);
}

__device__ __forceinline__ v8f wmma_bf16(v16bf a, v16bf b, v8f c) {
  return __builtin_amdgcn_wmma_f32_16x16x32_bf16(false, a, false, b, (short)0, c,
                                                 false, false);
}

// ---------------- prep: fp32 -> bf16 conversions ----------------
__global__ void cvt_bf16_kernel(const float* __restrict__ src,
                                uint16_t* __restrict__ dst, int n) {
  int i = blockIdx.x * blockDim.x + threadIdx.x;
  if (i < n) dst[i] = f2bf(src[i]);
}

// WT[n][k] = W[k][n]  (so B-fragments read contiguous K per lane)
__global__ void transpose_w_kernel(const float* __restrict__ w,
                                   uint16_t* __restrict__ wt) {
  int i = blockIdx.x * blockDim.x + threadIdx.x;
  if (i < En * En) {
    int n = i / En, k = i % En;
    wt[i] = f2bf(w[k * En + n]);
  }
}

// ---------------- QKV projection GEMM (bf16 WMMA) ----------------
// C[8192 x 768] = Xb[8192 x 768] * W[768 x 768] + bias
// 64x64 wave tile: 16 WMMA per 16 b128 loads per K-step (1 load / WMMA).
// blockIdx.z selects Q/K/V. Q scaled by 1/sqrt(D). Q,K stored [b,h,s,d]; V stored [b,h,d,s].
__global__ __launch_bounds__(128) void qkv_gemm_kernel(
    const uint16_t* __restrict__ Xb, const uint16_t* __restrict__ WTall,
    const float* __restrict__ bq, const float* __restrict__ bk,
    const float* __restrict__ bv, uint16_t* __restrict__ Qb,
    uint16_t* __restrict__ Kb, uint16_t* __restrict__ Vt) {
  const int sel = blockIdx.z;
  const uint16_t* WT = WTall + (size_t)sel * En * En;
  const float* bias = (sel == 0) ? bq : (sel == 1 ? bk : bv);

  const int wave = threadIdx.x >> 5, lane = threadIdx.x & 31;
  const int half = lane >> 4, l16 = lane & 15;
  const int wm = wave & 1, wn = wave >> 1;          // 2x2 waves per block
  const int mbase = blockIdx.x * 128 + wm * 64;     // token rows
  const int nbase = blockIdx.y * 128 + wn * 64;     // output cols

  FragC acc[4][4];
#pragma unroll
  for (int i = 0; i < 4; ++i)
#pragma unroll
    for (int j = 0; j < 4; ++j)
#pragma unroll
      for (int r = 0; r < 8; ++r) acc[i][j].f[r] = 0.0f;

  for (int k0 = 0; k0 < En; k0 += 32) {
    FragAB a[4], bfr[4];
#pragma unroll
    for (int i = 0; i < 4; ++i) {
      // A 16x32 bf16 layout: half 0 -> K 0..7 & 16..23, half 1 -> K 8..15 & 24..31
      const uint4* ap =
          (const uint4*)(Xb + (size_t)(mbase + i * 16 + l16) * En + k0);
      a[i].q[0] = ap[half];
      a[i].q[1] = ap[2 + half];
    }
#pragma unroll
    for (int j = 0; j < 4; ++j) {
      // B 32x16 bf16 layout: lane = col, half 0 -> K 0..7 & 16..23, half 1 -> K 8..15 & 24..31
      const uint4* bp =
          (const uint4*)(WT + (size_t)(nbase + j * 16 + l16) * En + k0);
      bfr[j].q[0] = bp[half];
      bfr[j].q[1] = bp[2 + half];
    }
#pragma unroll
    for (int i = 0; i < 4; ++i)
#pragma unroll
      for (int j = 0; j < 4; ++j)
        acc[i][j].v = wmma_bf16(a[i].v, bfr[j].v, acc[i][j].v);
  }

  // epilogue: C layout lane=col n, row m = half*8 + r
#pragma unroll
  for (int i = 0; i < 4; ++i)
#pragma unroll
    for (int j = 0; j < 4; ++j)
#pragma unroll
      for (int r = 0; r < 8; ++r) {
        int m = mbase + i * 16 + half * 8 + r;        // global token
        int ncol = nbase + j * 16 + l16;              // output column
        float val = acc[i][j].f[r] + bias[ncol];
        int bi = m >> 12, s = m & (Sn - 1);
        int h = ncol >> 6, d = ncol & 63;
        size_t bh = (size_t)bi * Hn + h;
        if (sel == 0) {
          val *= 0.125f;  // 1/sqrt(64)
          Qb[(bh * Sn + s) * Dn + d] = f2bf(val);
        } else if (sel == 1) {
          Kb[(bh * Sn + s) * Dn + d] = f2bf(val);
        } else {
          Vt[(bh * Dn + d) * Sn + s] = f2bf(val);  // transposed for PV B-frags
        }
      }
}

// ---------------- banded flash attention (bf16 WMMA) ----------------
// Each wave owns 32 queries (two 16-row M-tiles) and shares K/V fragments
// between them: per 32-key tile, 16 b128 loads feed 16 WMMAs.
__global__ __launch_bounds__(256) void attn_kernel(
    const uint16_t* __restrict__ Qb, const uint16_t* __restrict__ Kb,
    const uint16_t* __restrict__ Vt, float* __restrict__ out) {
  const int c = blockIdx.x, h = blockIdx.y, b = blockIdx.z;
  const int wave = threadIdx.x >> 5, lane = threadIdx.x & 31;
  const int half = lane >> 4, l16 = lane & 15;

  __shared__ __align__(16) uint16_t pbuf[8][16 * 32];  // per-wave P bounce

  const size_t bh = (size_t)b * Hn + h;
  const uint16_t* Qp = Qb + bh * Sn * Dn;  // [s][d]
  const uint16_t* Kp = Kb + bh * Sn * Dn;  // [s][d]
  const uint16_t* Vp = Vt + bh * Dn * Sn;  // [d][s]

  const int mbase0 = wave * 32;  // first query of this wave within the chunk

  // Q A-fragments for both M-tiles (d 0..31 / d 32..63), loaded once
  FragAB qa[2][2];
#pragma unroll
  for (int mt = 0; mt < 2; ++mt) {
    const uint4* qp =
        (const uint4*)(Qp + (size_t)(c * Wn + mbase0 + mt * 16 + l16) * Dn);
    qa[mt][0].q[0] = qp[half];
    qa[mt][0].q[1] = qp[2 + half];
    qa[mt][1].q[0] = qp[4 + half];
    qa[mt][1].q[1] = qp[6 + half];
  }

  float mrow[2][8], lrow[2][8];
  FragC acc[2][4];
#pragma unroll
  for (int mt = 0; mt < 2; ++mt)
#pragma unroll
    for (int r = 0; r < 8; ++r) {
      mrow[mt][r] = -1e30f;
      lrow[mt][r] = 0.0f;
    }
#pragma unroll
  for (int mt = 0; mt < 2; ++mt)
#pragma unroll
    for (int f = 0; f < 4; ++f)
#pragma unroll
      for (int r = 0; r < 8; ++r) acc[mt][f].f[r] = 0.0f;

  for (int kt = 0; kt < KT; ++kt) {
    const int tbase = kt * 32;             // t within [0, 3W)
    const int jlo = c * Wn - Wn + tbase;   // global key index of t = tbase
    // wave-uniform tile skips (band + sequence bounds) -> EXEC stays full
    if (tbase + 31 < mbase0) continue;           // rel < -W for all 32 queries
    if (tbase > mbase0 + 31 + 2 * Wn) continue;  // rel >  W for all 32 queries
    if (jlo + 31 < 0 || jlo > Sn - 1) continue;  // outside sequence

    // prefetch next key tile's K/V lines into WGP$ (global_prefetch_b8)
    {
      int jn = min(max(jlo + 32 + l16, 0), Sn - 1);
      __builtin_prefetch(Kp + (size_t)jn * Dn, 0, 0);
      int kn = min(max(jlo + 32 + half * 16, 0), Sn - 16);
      __builtin_prefetch(Vp + (size_t)l16 * Sn + kn, 0, 0);
    }

    // --- K^T B-fragments (shared by both M-tiles) ---
    FragAB kb[2][2];  // [key subtile][d step]
    {
      int r0 = min(max(jlo + l16, 0), Sn - 1);
      int r1 = min(max(jlo + 16 + l16, 0), Sn - 1);
      const uint4* k0 = (const uint4*)(Kp + (size_t)r0 * Dn);
      const uint4* k1 = (const uint4*)(Kp + (size_t)r1 * Dn);
      kb[0][0].q[0] = k0[half];     kb[0][0].q[1] = k0[2 + half];
      kb[0][1].q[0] = k0[4 + half]; kb[0][1].q[1] = k0[6 + half];
      kb[1][0].q[0] = k1[half];     kb[1][0].q[1] = k1[2 + half];
      kb[1][1].q[0] = k1[4 + half]; kb[1][1].q[1] = k1[6 + half];
    }
    // --- V B-fragments (shared by both M-tiles) ---
    FragAB vb[4];
    {
      int kst = min(max(jlo + half * 16, 0), Sn - 16);  // 16 contiguous keys
#pragma unroll
      for (int ds = 0; ds < 4; ++ds) {
        const uint4* vp =
            (const uint4*)(Vp + (size_t)(ds * 16 + l16) * Sn + kst);
        vb[ds].q[0] = vp[0];
        vb[ds].q[1] = vp[1];
      }
    }

    const int t0 = tbase + l16, t1 = tbase + 16 + l16;
    const int j0 = c * Wn + t0 - Wn, j1 = c * Wn + t1 - Wn;

#pragma unroll
    for (int mt = 0; mt < 2; ++mt) {
      const int mb = mbase0 + mt * 16;

      // --- scores: S = Q(16x64) * K^T(64x32) ---
      FragC sc0, sc1;
#pragma unroll
      for (int r = 0; r < 8; ++r) { sc0.f[r] = 0.0f; sc1.f[r] = 0.0f; }
      sc0.v = wmma_bf16(qa[mt][0].v, kb[0][0].v, sc0.v);
      sc0.v = wmma_bf16(qa[mt][1].v, kb[0][1].v, sc0.v);
      sc1.v = wmma_bf16(qa[mt][0].v, kb[1][0].v, sc1.v);
      sc1.v = wmma_bf16(qa[mt][1].v, kb[1][1].v, sc1.v);

      // --- mask + online softmax (row = half*8 + r, spread over 16 lanes) ---
#pragma unroll
      for (int r = 0; r < 8; ++r) {
        int p = mb + half * 8 + r;  // query offset in chunk
        int rel0 = t0 - Wn - p, rel1 = t1 - Wn - p;
        bool v0 = (rel0 >= -Wn) && (rel0 <= Wn) && (j0 >= 0) && (j0 < Sn);
        bool v1 = (rel1 >= -Wn) && (rel1 <= Wn) && (j1 >= 0) && (j1 < Sn);
        float x0 = v0 ? sc0.f[r] : -1e30f;
        float x1 = v1 ? sc1.f[r] : -1e30f;
        float tmax = fmaxf(x0, x1);
#pragma unroll
        for (int off = 1; off < 16; off <<= 1)
          tmax = fmaxf(tmax, __shfl_xor(tmax, off, 32));
        float nm = fmaxf(mrow[mt][r], tmax);
        float corr = __expf(mrow[mt][r] - nm);
        float e0 = v0 ? __expf(x0 - nm) : 0.0f;
        float e1 = v1 ? __expf(x1 - nm) : 0.0f;
        float rs = e0 + e1;
#pragma unroll
        for (int off = 1; off < 16; off <<= 1) rs += __shfl_xor(rs, off, 32);
        lrow[mt][r] = lrow[mt][r] * corr + rs;
        mrow[mt][r] = nm;
#pragma unroll
        for (int f = 0; f < 4; ++f) acc[mt][f].f[r] *= corr;
        // stash P (bf16) into LDS, plain row-major 16x32
        int row = half * 8 + r;
        pbuf[wave][row * 32 + l16] = f2bf(e0);
        pbuf[wave][row * 32 + 16 + l16] = f2bf(e1);
      }
      // DS ops are in-order per wave; fence keeps the compiler from reordering
      // and guarantees the stores landed before the A-layout reload.
      asm volatile("s_wait_dscnt 0" ::: "memory");

      // --- P(16x32) * V(32x64): A-frag from LDS, shared V B-frags ---
      FragAB pa;
      {
        const uint4* pp = (const uint4*)&pbuf[wave][l16 * 32];
        pa.q[0] = pp[2 * half];
        pa.q[1] = pp[2 * half + 1];
      }
#pragma unroll
      for (int ds = 0; ds < 4; ++ds)
        acc[mt][ds].v = wmma_bf16(pa.v, vb[ds].v, acc[mt][ds].v);
    }  // mt
  }  // key tiles

  // --- normalize + store out[b][s][h*64+d] fp32 ---
#pragma unroll
  for (int mt = 0; mt < 2; ++mt)
#pragma unroll
    for (int r = 0; r < 8; ++r) {
      float inv = (lrow[mt][r] > 0.0f) ? (1.0f / lrow[mt][r]) : 0.0f;
      int s = c * Wn + mbase0 + mt * 16 + half * 8 + r;
#pragma unroll
      for (int ds = 0; ds < 4; ++ds) {
        int d = ds * 16 + l16;
        out[((size_t)b * Sn + s) * En + h * Dn + d] = acc[mt][ds].f[r] * inv;
      }
    }
}

// ---------------- host-side launcher ----------------
extern "C" void kernel_launch(void* const* d_in, const int* in_sizes, int n_in,
                              void* d_out, int out_size, void* d_ws,
                              size_t ws_size, hipStream_t stream) {
  (void)in_sizes; (void)n_in; (void)out_size;
  const float* hs = (const float*)d_in[0];
  const float* Wq = (const float*)d_in[1];
  const float* bq = (const float*)d_in[2];
  const float* Wk = (const float*)d_in[3];
  const float* bk = (const float*)d_in[4];
  const float* Wv = (const float*)d_in[5];
  const float* bv = (const float*)d_in[6];
  float* out = (float*)d_out;

  char* ws = (char*)d_ws;
  size_t off = 0;
  const size_t xElems = (size_t)Bn * Sn * En;        // 6.29M
  const size_t wElems = (size_t)En * En;             // 589824
  const size_t qkvElems = (size_t)Bn * Hn * Sn * Dn; // 6.29M
  uint16_t* Xb = (uint16_t*)(ws + off); off += xElems * 2;
  uint16_t* WT = (uint16_t*)(ws + off); off += 3 * wElems * 2;
  uint16_t* Qb = (uint16_t*)(ws + off); off += qkvElems * 2;
  uint16_t* Kb = (uint16_t*)(ws + off); off += qkvElems * 2;
  uint16_t* Vt = (uint16_t*)(ws + off); off += qkvElems * 2;
  if (ws_size < off) return;  // need ~54 MB of scratch

  int nx = (int)xElems;
  cvt_bf16_kernel<<<(nx + 255) / 256, 256, 0, stream>>>(hs, Xb, nx);
  int nw = (int)wElems;
  transpose_w_kernel<<<(nw + 255) / 256, 256, 0, stream>>>(Wq, WT);
  transpose_w_kernel<<<(nw + 255) / 256, 256, 0, stream>>>(Wk, WT + wElems);
  transpose_w_kernel<<<(nw + 255) / 256, 256, 0, stream>>>(Wv, WT + 2 * wElems);

  dim3 gGemm((Bn * Sn) / 128, En / 128, 3);  // 64 x 6 x 3, 128 threads
  qkv_gemm_kernel<<<gGemm, 128, 0, stream>>>(Xb, WT, bq, bk, bv, Qb, Kb, Vt);

  dim3 gAttn(Cn, Hn, Bn);  // 16 x 12 x 2
  attn_kernel<<<gAttn, 256, 0, stream>>>(Qb, Kb, Vt, out);
}
